// DecoderBlock_29944511988288
// MI455X (gfx1250) — matrix operand order repaired and checked by reading, not compiled, not run on hardware
//
#include <hip/hip_runtime.h>
#include <hip/hip_bf16.h>

#define S_LEN 2048
#define EMB   2048
#define NH    16
#define HD    128            // EMB / NH
#define LD3   (3 * EMB)

typedef __attribute__((ext_vector_type(16))) __bf16 v16bf;
typedef __attribute__((ext_vector_type(8)))  __bf16 v8bf;
typedef __attribute__((ext_vector_type(8)))  float  v8f;
typedef __attribute__((ext_vector_type(4)))  int    v4i;

union AFrag { v16bf v; v8bf h[2]; };

__device__ __forceinline__ v8f wmma_bf16(v16bf a, v16bf b, v8f c) {
  // (neg_a, A, neg_b, B, c_mod, C, reuse_a, reuse_b)
  return __builtin_amdgcn_wmma_f32_16x16x32_bf16(false, a, false, b, (short)0, c, false, false);
}

// A operand (16x32 bf16) from a row-major [M,K] tile: lane = row L%16,
// K chunks (L/16)*8 and 16+(L/16)*8 (per CDNA5 ISA 16-bit A layout).
__device__ __forceinline__ v16bf load_a(const __bf16* base, int ld, int k0, int lane) {
  const int row = lane & 15, hi = lane >> 4;
  const __bf16* p = base + (size_t)row * ld + k0 + hi * 8;
  AFrag f;
  f.h[0] = *(const v8bf*)(p);
  f.h[1] = *(const v8bf*)(p + 16);
  return f.v;
}

// B operand (32x16 bf16) taken from row-major [N,K] storage (i.e. B^T):
// lane = col L%16, K chunks (L/16)*8 and 16+(L/16)*8 (matches load_a).
__device__ __forceinline__ v16bf load_b(const __bf16* base, int ld, int k0, int lane) {
  const int col = lane & 15, hi = lane >> 4;
  const __bf16* p = base + (size_t)col * ld + k0 + hi * 8;
  AFrag f;
  f.h[0] = *(const v8bf*)(p);
  f.h[1] = *(const v8bf*)(p + 16);
  return f.v;
}

// ---- CDNA5 async global->LDS staging (ASYNCcnt path), guarded fallback ----
#if __has_builtin(__builtin_amdgcn_global_load_async_to_lds_b128)
#define HAVE_ASYNC_LDS 1
#else
#define HAVE_ASYNC_LDS 0
#endif

typedef __attribute__((address_space(1))) v4i* gas_b128_ptr;   // global 128b payload
typedef __attribute__((address_space(3))) v4i* las_b128_ptr;   // LDS    128b payload

__device__ __forceinline__ void async_copy_b128(const __bf16* g, __bf16* l) {
#if HAVE_ASYNC_LDS
  __builtin_amdgcn_global_load_async_to_lds_b128(
      (gas_b128_ptr)(__bf16*)g,      // cast away const, then generic->AS1
      (las_b128_ptr)l,               // generic->AS3
      0, 0);
#else
  *(v8bf*)l = *(const v8bf*)g;       // sync bounce through VGPRs
#endif
}

__device__ __forceinline__ void wait_async_copies() {
#if HAVE_ASYNC_LDS
#if __has_builtin(__builtin_amdgcn_s_wait_asynccnt)
  __builtin_amdgcn_s_wait_asynccnt(0);
#else
  asm volatile("s_wait_asynccnt 0" ::: "memory");
#endif
#endif
}

// ---------------------------------------------------------------------------
// Generic GEMM: C[M,N] = A[M,K] * B[N,K]^T  (bf16 operands, fp32 accum).
// Block tile 128x128, K-slab 64, double-buffered LDS (64KB) filled with
// async global->LDS b128 copies. 8 waves (4x2); wave tile 32x64 = 2x4 WMMA.
// EPI 0: store bf16. EPI 1: out = resid + C (fp32). EPI 2: gelu(C) -> bf16.
// ---------------------------------------------------------------------------
template<int EPI>
__global__ __launch_bounds__(256) void gemm_bf16_nt(
    const __bf16* __restrict__ A, const __bf16* __restrict__ B,
    int M, int N, int K,
    const float* __restrict__ resid, float* __restrict__ outF,
    __bf16* __restrict__ outB)
{
  __shared__ __bf16 ldsA[2][128 * 64];
  __shared__ __bf16 ldsB[2][128 * 64];

  const int lane = threadIdx.x & 31;
  const int wave = threadIdx.x >> 5;
  const int wm = wave & 3;                 // 4 waves along M (32 rows each)
  const int wn = wave >> 2;                // 2 waves along N (64 cols each)
  const int mblk = blockIdx.y * 128;
  const int nblk = blockIdx.x * 128;
  const __bf16* Ag = A + (size_t)mblk * K;
  const __bf16* Bg = B + (size_t)nblk * K;

  // Stage one 128x64 A-slab + 128x64 B-slab (16KB each) into LDS buffer b.
  auto stage = [&](int k0, int b) {
    #pragma unroll
    for (int it = 0; it < 4; ++it) {
      const int i   = threadIdx.x + it * 256;     // 0..1023
      const int row = i >> 3;
      const int ch  = (i & 7) * 8;                // 8 bf16 = 16B chunks
      async_copy_b128(Ag + (size_t)row * K + k0 + ch, &ldsA[b][row * 64 + ch]);
      async_copy_b128(Bg + (size_t)row * K + k0 + ch, &ldsB[b][row * 64 + ch]);
    }
  };

  v8f c[2][4] = {};
  stage(0, 0);
  int buf = 0;
  for (int k0 = 0; k0 < K; k0 += 64) {
    wait_async_copies();
    __syncthreads();                         // slab `buf` visible to all waves
    if (k0 + 64 < K) stage(k0 + 64, buf ^ 1);

    const __bf16* la = &ldsA[buf][(wm * 32) * 64];
    const __bf16* lb = &ldsB[buf][(wn * 64) * 64];
    #pragma unroll
    for (int kk = 0; kk < 64; kk += 32) {
      v16bf a0 = load_a(la,           64, kk, lane);
      v16bf a1 = load_a(la + 16 * 64, 64, kk, lane);
      v16bf b0 = load_b(lb,           64, kk, lane);
      v16bf b1 = load_b(lb + 16 * 64, 64, kk, lane);
      v16bf b2 = load_b(lb + 32 * 64, 64, kk, lane);
      v16bf b3 = load_b(lb + 48 * 64, 64, kk, lane);
      c[0][0] = wmma_bf16(a0, b0, c[0][0]);
      c[0][1] = wmma_bf16(a0, b1, c[0][1]);
      c[0][2] = wmma_bf16(a0, b2, c[0][2]);
      c[0][3] = wmma_bf16(a0, b3, c[0][3]);
      c[1][0] = wmma_bf16(a1, b0, c[1][0]);
      c[1][1] = wmma_bf16(a1, b1, c[1][1]);
      c[1][2] = wmma_bf16(a1, b2, c[1][2]);
      c[1][3] = wmma_bf16(a1, b3, c[1][3]);
    }
    __syncthreads();                         // done reading `buf`
    buf ^= 1;
  }

  const int col = lane & 15, hi = lane >> 4;
  const int m0 = mblk + wm * 32;
  const int n0 = nblk + wn * 64;
  #pragma unroll
  for (int i = 0; i < 2; ++i) {
    #pragma unroll
    for (int j = 0; j < 4; ++j) {
      #pragma unroll
      for (int r = 0; r < 8; ++r) {
        const int gr = m0 + i * 16 + hi * 8 + r;
        const int gc = n0 + j * 16 + col;
        const size_t idx = (size_t)gr * N + gc;
        const float v = c[i][j][r];
        if constexpr (EPI == 0) {
          outB[idx] = (__bf16)v;
        } else if constexpr (EPI == 1) {
          outF[idx] = resid[idx] + v;
        } else {
          const float t = tanhf(0.7978845608028654f * (v + 0.044715f * v * v * v));
          outB[idx] = (__bf16)(0.5f * v * (1.0f + t));
        }
      }
    }
  }
}

// ---------------------------------------------------------------------------
// LayerNorm (no bias): fp32 in -> bf16 out, one block (256 thr) per row.
// ---------------------------------------------------------------------------
__global__ __launch_bounds__(256) void layernorm_bf16(
    const float* __restrict__ x, const float* __restrict__ w, __bf16* __restrict__ out)
{
  const int row = blockIdx.x;
  const float* xr = x + (size_t)row * EMB;
  float v[8], s1 = 0.f, s2 = 0.f;
  #pragma unroll
  for (int i = 0; i < 8; ++i) {
    const float t = xr[threadIdx.x + i * 256];
    v[i] = t; s1 += t; s2 += t * t;
  }
  #pragma unroll
  for (int m = 1; m < 32; m <<= 1) { s1 += __shfl_xor(s1, m); s2 += __shfl_xor(s2, m); }
  __shared__ float a1[8], a2[8];
  const int wv = threadIdx.x >> 5, ln = threadIdx.x & 31;
  if (ln == 0) { a1[wv] = s1; a2[wv] = s2; }
  __syncthreads();
  float t1 = 0.f, t2 = 0.f;
  #pragma unroll
  for (int i = 0; i < 8; ++i) { t1 += a1[i]; t2 += a2[i]; }
  const float mu   = t1 * (1.0f / EMB);
  const float var  = t2 * (1.0f / EMB) - mu * mu;
  const float rstd = rsqrtf(var + 1e-5f);
  #pragma unroll
  for (int i = 0; i < 8; ++i) {
    const int c = threadIdx.x + i * 256;
    out[(size_t)row * EMB + c] = (__bf16)((v[i] - mu) * rstd * w[c]);
  }
}

__global__ void cvt_f32_bf16(const float* __restrict__ in, __bf16* __restrict__ out, int n) {
  for (int i = blockIdx.x * blockDim.x + threadIdx.x; i < n; i += gridDim.x * blockDim.x)
    out[i] = (__bf16)in[i];
}

// vT[e*S + s] = qkv[s*3E + 2E + e]   (V part of fused QKV, transposed per head)
__global__ __launch_bounds__(256) void v_transpose(
    const __bf16* __restrict__ qkv, __bf16* __restrict__ vT)
{
  __shared__ __bf16 tile[32][33];
  const int tx = threadIdx.x & 31, ty = threadIdx.x >> 5;   // 32 x 8
  const int s0 = blockIdx.y * 32, e0 = blockIdx.x * 32;
  #pragma unroll
  for (int i = 0; i < 32; i += 8)
    tile[ty + i][tx] = qkv[(size_t)(s0 + ty + i) * LD3 + 2 * EMB + e0 + tx];
  __syncthreads();
  #pragma unroll
  for (int i = 0; i < 32; i += 8)
    vT[(size_t)(e0 + ty + i) * S_LEN + s0 + tx] = tile[tx][ty + i];
}

// ---------------------------------------------------------------------------
// Flash-style causal attention. Grid (S/64, H), block 128 = 4 waves.
// Each wave: 16 query rows. Q kept as 4 A-frags; O as 8 fp32 C-frags.
// Key tiles of 32; online softmax with half-wave shfl reductions; P is
// re-laid C->A via a 1KB LDS bounce per wave.
// ---------------------------------------------------------------------------
__global__ __launch_bounds__(128) void attn_flash(
    const __bf16* __restrict__ qkv,   // [S, 3E]
    const __bf16* __restrict__ vT,    // [H][D][S]
    __bf16* __restrict__ attn_out)    // [S, E]
{
  const int h    = blockIdx.y;
  const int wave = threadIdx.x >> 5;
  const int lane = threadIdx.x & 31;
  const int q0   = blockIdx.x * 64 + wave * 16;
  const int col  = lane & 15, hi = lane >> 4;

  const __bf16* Qp = qkv + (size_t)h * HD;
  const __bf16* Kp = qkv + EMB + (size_t)h * HD;
  const __bf16* Vt = vT + (size_t)h * HD * S_LEN;

  __shared__ __bf16 lds_p[4][16][32];

  v16bf qf[4];
  #pragma unroll
  for (int kk = 0; kk < 4; ++kk)
    qf[kk] = load_a(Qp + (size_t)q0 * LD3, LD3, kk * 32, lane);

  v8f o[8] = {};
  float mst[8], lst[8];
  #pragma unroll
  for (int r = 0; r < 8; ++r) { mst[r] = -3.0e38f; lst[r] = 0.f; }

  const float scale = 0.08838834764831845f;   // 1/sqrt(128)
  const int ntiles = q0 / 32 + 1;             // causal bound: keys <= q0+15

  for (int t = 0; t < ntiles; ++t) {
    const int kb = t * 32;
    // ---- S = Q K^T (16x32 scores as two C-frags) ----
    v8f s0 = {}, s1 = {};
    #pragma unroll
    for (int kk = 0; kk < 4; ++kk) {
      v16bf b0 = load_b(Kp + (size_t)kb * LD3,        LD3, kk * 32, lane);
      v16bf b1 = load_b(Kp + (size_t)(kb + 16) * LD3, LD3, kk * 32, lane);
      s0 = wmma_bf16(qf[kk], b0, s0);
      s1 = wmma_bf16(qf[kk], b1, s1);
    }
    // ---- scale + causal bias + online softmax ----
    float alpha[8];
    #pragma unroll
    for (int r = 0; r < 8; ++r) {
      const int qi = q0 + hi * 8 + r;
      float v0 = s0[r] * scale + ((kb + col)      > qi ? -10000.f : 0.f);
      float v1 = s1[r] * scale + ((kb + 16 + col) > qi ? -10000.f : 0.f);
      float mloc = fmaxf(v0, v1);
      #pragma unroll
      for (int msk = 1; msk < 16; msk <<= 1) mloc = fmaxf(mloc, __shfl_xor(mloc, msk));
      const float mnew = fmaxf(mst[r], mloc);
      const float a    = __expf(mst[r] - mnew);
      const float p0   = __expf(v0 - mnew);
      const float p1   = __expf(v1 - mnew);
      float ls = p0 + p1;
      #pragma unroll
      for (int msk = 1; msk < 16; msk <<= 1) ls += __shfl_xor(ls, msk);
      lst[r] = lst[r] * a + ls;
      mst[r] = mnew;
      alpha[r] = a;
      lds_p[wave][hi * 8 + r][col]      = (__bf16)p0;
      lds_p[wave][hi * 8 + r][16 + col] = (__bf16)p1;
    }
    #pragma unroll
    for (int c = 0; c < 8; ++c)
      #pragma unroll
      for (int r = 0; r < 8; ++r) o[c][r] *= alpha[r];
    __builtin_amdgcn_wave_barrier();          // same-wave LDS RAW; DS is in-order
    v16bf pa = load_a(&lds_p[wave][0][0], 32, 0, lane);
    __builtin_amdgcn_wave_barrier();
    // ---- O += P V ----
    #pragma unroll
    for (int c = 0; c < 8; ++c) {
      v16bf vb = load_b(Vt + (size_t)(c * 16) * S_LEN, S_LEN, kb, lane);
      o[c] = wmma_bf16(pa, vb, o[c]);
    }
  }
  // ---- O / l -> bf16 ----
  #pragma unroll
  for (int c = 0; c < 8; ++c)
    #pragma unroll
    for (int r = 0; r < 8; ++r) {
      const int qi = q0 + hi * 8 + r;
      attn_out[(size_t)qi * EMB + h * HD + c * 16 + col] = (__bf16)(o[c][r] / lst[r]);
    }
}

// ---------------------------------------------------------------------------
extern "C" void kernel_launch(void* const* d_in, const int* in_sizes, int n_in,
                              void* d_out, int out_size, void* d_ws, size_t ws_size,
                              hipStream_t stream) {
  const float* x       = (const float*)d_in[0];
  const float* ln1_w   = (const float*)d_in[1];
  const float* ln2_w   = (const float*)d_in[2];
  const float* w_attn  = (const float*)d_in[3];
  const float* w_projA = (const float*)d_in[4];
  const float* w_fc    = (const float*)d_in[5];
  const float* w_projF = (const float*)d_in[6];

  char* p = (char*)d_ws;
  auto alloc = [&](size_t bytes) { void* r = (void*)p; p += (bytes + 255) & ~(size_t)255; return r; };
  __bf16* wqkv_b = (__bf16*)alloc((size_t)3 * EMB * EMB * 2);
  __bf16* wprA_b = (__bf16*)alloc((size_t)EMB * EMB * 2);
  __bf16* wfc_b  = (__bf16*)alloc((size_t)4 * EMB * EMB * 2);
  __bf16* wprF_b = (__bf16*)alloc((size_t)4 * EMB * EMB * 2);
  __bf16* hb     = (__bf16*)alloc((size_t)S_LEN * EMB * 2);
  __bf16* qkv_b  = (__bf16*)alloc((size_t)S_LEN * 3 * EMB * 2);
  __bf16* vT     = (__bf16*)alloc((size_t)EMB * S_LEN * 2);
  __bf16* attn_b = (__bf16*)alloc((size_t)S_LEN * EMB * 2);
  float*  x1     = (float*) alloc((size_t)S_LEN * EMB * 4);
  __bf16* h2b    = (__bf16*)alloc((size_t)S_LEN * EMB * 2);
  __bf16* gb     = (__bf16*)alloc((size_t)S_LEN * 4 * EMB * 2);

  cvt_f32_bf16<<<2048, 256, 0, stream>>>(w_attn,  wqkv_b, 3 * EMB * EMB);
  cvt_f32_bf16<<<2048, 256, 0, stream>>>(w_projA, wprA_b, EMB * EMB);
  cvt_f32_bf16<<<2048, 256, 0, stream>>>(w_fc,    wfc_b,  4 * EMB * EMB);
  cvt_f32_bf16<<<2048, 256, 0, stream>>>(w_projF, wprF_b, 4 * EMB * EMB);

  layernorm_bf16<<<S_LEN, 256, 0, stream>>>(x, ln1_w, hb);

  gemm_bf16_nt<0><<<dim3(3 * EMB / 128, S_LEN / 128), 256, 0, stream>>>(
      hb, wqkv_b, S_LEN, 3 * EMB, EMB, nullptr, nullptr, qkv_b);

  v_transpose<<<dim3(EMB / 32, S_LEN / 32), 256, 0, stream>>>(qkv_b, vT);

  attn_flash<<<dim3(S_LEN / 64, NH), 128, 0, stream>>>(qkv_b, vT, attn_b);

  gemm_bf16_nt<1><<<dim3(EMB / 128, S_LEN / 128), 256, 0, stream>>>(
      attn_b, wprA_b, S_LEN, EMB, EMB, x, x1, nullptr);

  layernorm_bf16<<<S_LEN, 256, 0, stream>>>(x1, ln2_w, h2b);

  gemm_bf16_nt<2><<<dim3(4 * EMB / 128, S_LEN / 128), 256, 0, stream>>>(
      h2b, wfc_b, S_LEN, 4 * EMB, EMB, nullptr, nullptr, gb);

  gemm_bf16_nt<1><<<dim3(EMB / 128, S_LEN / 128), 256, 0, stream>>>(
      gb, wprF_b, S_LEN, EMB, 4 * EMB, x1, (float*)d_out, nullptr);

  (void)in_sizes; (void)n_in; (void)out_size; (void)ws_size;
}
